// L2Attention_35983236006170
// MI455X (gfx1250) — matrix operand order repaired and checked by reading, not compiled, not run on hardware
//
#include <hip/hip_runtime.h>
#include <math.h>

// Problem constants (fixed by the reference)
#define Bn 4
#define Cc 128
#define Nn 4096
#define Kk 32
#define Hh 4
#define Dd 32
#define Ff 512

typedef __bf16 bf16;
typedef __attribute__((ext_vector_type(16))) bf16 v16bf;
typedef __attribute__((ext_vector_type(2)))  bf16 bf16x2;
typedef __attribute__((ext_vector_type(8)))  float v8f;

union FragU { v16bf v; bf16x2 h[8]; };

__device__ __forceinline__ bf16x2 pack2f(float f0, float f1) {
  bf16x2 r; r[0] = (bf16)f0; r[1] = (bf16)f1; return r;
}

// A fragment (16x32 bf16) from a row-major [row][k] bf16 buffer.
// ISA layout: lanes 0-15 row M=lane hold K 0-7 (v0..3) & 16-23 (v4..7);
// lanes 16-31 hold K 8-15 & 24-31. Each VGPR = 2 packed consecutive K.
__device__ __forceinline__ v16bf load_a_bf16(const bf16* base, int strideH,
                                             int row, int hi, int kt) {
  FragU u;
#pragma unroll
  for (int i = 0; i < 8; ++i) {
    int kb = kt * 32 + ((i & 4) ? 16 : 0) + hi * 8 + (i & 3) * 2;
    u.h[i] = *reinterpret_cast<const bf16x2*>(base + row * strideH + kb);
  }
  return u.v;
}

// B fragment (32x16 bf16) from a [col][k] bf16 buffer.
// Layout: col N = lane&15; lanes 0-15 hold K 0-7 & 16-23, lanes 16-31 K 8-15 & 24-31,
// packed 2 consecutive K per VGPR.
__device__ __forceinline__ v16bf load_b_bf16(const bf16* base, int strideH,
                                             int col, int hi, int kt) {
  FragU u;
#pragma unroll
  for (int i = 0; i < 8; ++i) {
    int kb = kt * 32 + ((i & 4) ? 16 : 0) + hi * 8 + (i & 3) * 2;
    u.h[i] = *reinterpret_cast<const bf16x2*>(base + col * strideH + kb);
  }
  return u.v;
}

// A fragment from an fp32 row-major [row][k] buffer (global or LDS),
// bf16-converted on the fly.
__device__ __forceinline__ v16bf load_a_f32(const float* base, int stride,
                                            int row, int hi, int kt) {
  FragU u;
#pragma unroll
  for (int i = 0; i < 8; ++i) {
    int kb = kt * 32 + ((i & 4) ? 16 : 0) + hi * 8 + (i & 3) * 2;
    const float* p = base + row * stride + kb;
    u.h[i] = pack2f(p[0], p[1]);
  }
  return u.v;
}

// B fragment from an fp32 [col][k] buffer, bf16-converted on the fly.
__device__ __forceinline__ v16bf load_b_f32(const float* base, int stride,
                                            int col, int hi, int kt) {
  FragU u;
#pragma unroll
  for (int i = 0; i < 8; ++i) {
    int kb = kt * 32 + ((i & 4) ? 16 : 0) + hi * 8 + (i & 3) * 2;
    const float* p = base + col * stride + kb;
    u.h[i] = pack2f(p[0], p[1]);
  }
  return u.v;
}

__device__ __forceinline__ v8f wmma_bf16(v16bf a, v16bf b, v8f c) {
  return __builtin_amdgcn_wmma_f32_16x16x32_bf16(false, a, false, b,
                                                 (short)0, c, false, false);
}

// CDNA5 async global->LDS copy (ASYNCcnt). ldsOff is the LDS byte address:
// with purely dynamic LDS (no static __shared__) the dynamic block starts at
// LDS offset 0, so plain integer offsets are valid.
__device__ __forceinline__ void async_copy_b32(unsigned ldsOff,
                                               const void* gptr) {
  asm volatile("global_load_async_to_lds_b32 %0, %1, off"
               :: "v"(ldsOff), "v"((unsigned long long)(uintptr_t)gptr)
               : "memory");
}
__device__ __forceinline__ void async_copy_b128(unsigned ldsOff,
                                                const void* gptr) {
  asm volatile("global_load_async_to_lds_b128 %0, %1, off"
               :: "v"(ldsOff), "v"((unsigned long long)(uintptr_t)gptr)
               : "memory");
}
__device__ __forceinline__ void wait_async0() {
  asm volatile("s_wait_asynccnt 0x0" ::: "memory");
}

// ---------------------------------------------------------------------------
// Kernel 1: Q/K/V projections. Exploits linearity: K,V convs are applied to x
// once; neighbor-diffs are formed later from gathered projections.
// x tile and W tiles are staged to LDS with CDNA5 async copies (fp32; the
// dword-granular per-lane LDS addressing performs the x transpose for free),
// and converted to bf16 during fragment assembly.
// Outputs in (B,N,C) so per-(b,n) channel slices are contiguous for gathers.
// Also emits x in bf16 (B,N,C) for the KNN gram matmul.
// grid (N/128, B), block 256 (8 waves). dyn LDS = 2*128*132*4 bytes.
// ---------------------------------------------------------------------------
__global__ void qkv_kernel(const float* __restrict__ x,
                           const float* __restrict__ Wq,
                           const float* __restrict__ Wk,
                           const float* __restrict__ Wv,
                           float* __restrict__ q, float* __restrict__ kx,
                           float* __restrict__ vx, bf16* __restrict__ xb) {
  extern __shared__ char smem[];
  float* xlds = (float*)smem;                    // [128 n][132] fp32, off 0
  float* wlds = (float*)(smem + 128 * 132 * 4);  // [128 co][132] fp32
  const unsigned wldsOff = 128u * 132u * 4u;
  const int tid = threadIdx.x;
  const int b = blockIdx.y;
  const int n0 = blockIdx.x * 128;

  // Async-stage x tile transposed to [n][c] (coalesced over n per channel;
  // the transpose happens via the per-lane LDS destination address).
  for (int e = tid; e < 128 * 128; e += 256) {
    int c = e >> 7, nl = e & 127;
    const float* gp = x + ((size_t)(b * Cc + c)) * Nn + n0 + nl;
    async_copy_b32((unsigned)((nl * 132 + c) * 4), gp);
  }
  wait_async0();
  __syncthreads();

  // Persist bf16 x row-major (B,N,C) for the gram kernel (coalesced over c).
  for (int e = tid; e < 128 * 128; e += 256) {
    int nl = e >> 7, c = e & 127;
    xb[((size_t)(b * Nn + n0 + nl)) * Cc + c] = (bf16)xlds[nl * 132 + c];
  }

  const int wave = tid >> 5, lane = tid & 31;
  const int r = lane & 15, hi = lane >> 4;
  const int nl0 = wave * 16;

  v16bf afrag[4];
#pragma unroll
  for (int kt = 0; kt < 4; ++kt)
    afrag[kt] = load_a_f32(xlds, 132, nl0 + r, hi, kt);

  const float* Ws[3] = {Wq, Wk, Wv};
  float* Os[3] = {q, kx, vx};
#pragma unroll 1
  for (int wi = 0; wi < 3; ++wi) {
    const float* W = Ws[wi];
    float* O = Os[wi];
    __syncthreads();
    // Async-stage W (co, c) row-major, 16B-granular.
    for (int e = tid; e < 128 * 32; e += 256) {
      int co = e >> 5, c4 = (e & 31) * 4;
      async_copy_b128(wldsOff + (unsigned)((co * 132 + c4) * 4),
                      W + co * Cc + c4);
    }
    wait_async0();
    __syncthreads();
#pragma unroll 1
    for (int ct = 0; ct < 8; ++ct) {
      int cb = ct * 16;
      v8f acc = {0, 0, 0, 0, 0, 0, 0, 0};
#pragma unroll
      for (int kt = 0; kt < 4; ++kt) {
        v16bf bfrag = load_b_f32(wlds, 132, cb + r, hi, kt);
        acc = wmma_bf16(afrag[kt], bfrag, acc);
      }
#pragma unroll
      for (int i = 0; i < 8; ++i) {
        int nrow = n0 + nl0 + i + 8 * hi;
        O[((size_t)(b * Nn + nrow)) * Cc + cb + r] = acc[i];
      }
    }
  }
}

// ---------------------------------------------------------------------------
// Kernel 2: exact fp32 squared norms per point. grid (B*N/256), block 256.
// ---------------------------------------------------------------------------
__global__ void sqnorm_kernel(const float* __restrict__ x,
                              float* __restrict__ sq) {
  int gid = blockIdx.x * 256 + threadIdx.x;  // = b*N + n
  int b = gid >> 12, n = gid & (Nn - 1);
  float s = 0.0f;
#pragma unroll 4
  for (int c = 0; c < Cc; ++c) {
    float v = x[((size_t)(b * Cc + c)) * Nn + n];
    s += v * v;
  }
  sq[gid] = s;
}

// ---------------------------------------------------------------------------
// Kernel 3: fused distance tile + top-32 selection.
// 16 rows per workgroup; 16x4096 fp32 distance tile lives entirely in the
// 320KB WGP LDS. Phase 1: WMMA gram (-2*x.xT) + fp32 norms. Phase 2: one wave
// per row does 32 argmin rounds with stripe-owned invalidation.
// grid (N/16, B), block 512 (16 waves). dyn LDS = 16*4096*4 = 256 KB.
// ---------------------------------------------------------------------------
__global__ void knn_kernel(const bf16* __restrict__ xb,
                           const float* __restrict__ sq,
                           int* __restrict__ idx) {
  extern __shared__ char smem[];
  float* dist = (float*)smem;  // [16][4096]
  const int tid = threadIdx.x;
  const int b = blockIdx.y;
  const int nr0 = blockIdx.x * 16;
  const int wave = tid >> 5, lane = tid & 31;
  const int r = lane & 15, hi = lane >> 4;
  const bf16* xbB = xb + (size_t)b * Nn * Cc;

  v16bf afrag[4];
#pragma unroll
  for (int kt = 0; kt < 4; ++kt)
    afrag[kt] = load_a_bf16(xbB, Cc, nr0 + r, hi, kt);

  float srow[8];
#pragma unroll
  for (int i = 0; i < 8; ++i) srow[i] = sq[b * Nn + nr0 + i + 8 * hi];

#pragma unroll 1
  for (int j = 0; j < 16; ++j) {
    int cb = (wave + 16 * j) * 16;  // 16 waves cover all 256 column tiles
    v8f acc = {0, 0, 0, 0, 0, 0, 0, 0};
#pragma unroll
    for (int kt = 0; kt < 4; ++kt) {
      v16bf bfrag = load_b_bf16(xbB, Cc, cb + r, hi, kt);
      acc = wmma_bf16(afrag[kt], bfrag, acc);
    }
    float sqc = sq[b * Nn + cb + r];
#pragma unroll
    for (int i = 0; i < 8; ++i)
      dist[(i + 8 * hi) * Nn + cb + r] = srow[i] + sqc - 2.0f * acc[i];
  }
  __syncthreads();

  // Phase 2: top-32 smallest per row; wave w owns row w.
  float* drow = dist + wave * Nn;
  int* orow = idx + ((size_t)(b * Nn + nr0 + wave)) * Kk;
#pragma unroll 1
  for (int sel = 0; sel < Kk; ++sel) {
    float best = 3.0e38f;
    int bi = 0;
#pragma unroll 4
    for (int jj = 0; jj < Nn / 32; ++jj) {
      int col = jj * 32 + lane;  // lane-striped: element owned by lane col&31
      float v = drow[col];
      if (v < best) { best = v; bi = col; }
    }
#pragma unroll
    for (int off = 16; off > 0; off >>= 1) {
      float ob = __shfl_xor(best, off, 32);
      int oi = __shfl_xor(bi, off, 32);
      if (ob < best || (ob == best && oi < bi)) { best = ob; bi = oi; }
    }
    if (lane == 0) orow[sel] = bi;
    // Invalidate chosen element; only its stripe-owner lane rereads it.
    if ((bi & 31) == lane) drow[bi] = 3.0e38f;
  }
}

// ---------------------------------------------------------------------------
// Kernel 4: vector attention + residual. One wave per (b,h,n), lane = d.
// k,v neighbor-diffs formed by gathering the *projections* (linearity).
// Softmax is over the D axis == over the 32 lanes of the wave.
// Neighbor rows are prefetched 2 iterations ahead (global_prefetch_b8) to
// hide the data-dependent gather latency behind the softmax VALU work.
// grid (N, B), block 128 (4 waves = 4 heads).
// ---------------------------------------------------------------------------
__global__ void attn_kernel(const float* __restrict__ x,
                            const float* __restrict__ q,
                            const float* __restrict__ kx,
                            const float* __restrict__ vx,
                            const int* __restrict__ idx,
                            float* __restrict__ y1) {
  const int n = blockIdx.x, b = blockIdx.y;
  const int c = threadIdx.x;  // c = h*32 + d ; wave = head, lane = d
  const size_t base = ((size_t)(b * Nn + n)) * Cc;
  const float qv = q[base + c];
  const float kc = kx[base + c];
  const float vc = vx[base + c];
  const int* ip = idx + ((size_t)(b * Nn + n)) * Kk;
  float acc = 0.0f;
  const float scale = 0.17677669529663687f;  // 1/sqrt(D=32)
  {  // prefetch first two neighbor rows
    size_t p0 = ((size_t)(b * Nn + ip[0])) * Cc + c;
    size_t p1 = ((size_t)(b * Nn + ip[1])) * Cc + c;
    __builtin_prefetch(kx + p0, 0, 0);
    __builtin_prefetch(vx + p0, 0, 0);
    __builtin_prefetch(kx + p1, 0, 0);
    __builtin_prefetch(vx + p1, 0, 0);
  }
#pragma unroll 1
  for (int k = 0; k < Kk; ++k) {
    if (k + 2 < Kk) {
      size_t pb = ((size_t)(b * Nn + ip[k + 2])) * Cc + c;
      __builtin_prefetch(kx + pb, 0, 0);
      __builtin_prefetch(vx + pb, 0, 0);
    }
    int m = ip[k];
    size_t mb = ((size_t)(b * Nn + m)) * Cc;
    float kd = kx[mb + c] - kc;
    float vd = vx[mb + c] - vc;
    float e = (qv - kd) * scale;
    float mx = e;
#pragma unroll
    for (int off = 16; off > 0; off >>= 1)
      mx = fmaxf(mx, __shfl_xor(mx, off, 32));
    float p = __expf(e - mx);
    float s = p;
#pragma unroll
    for (int off = 16; off > 0; off >>= 1) s += __shfl_xor(s, off, 32);
    acc += (p / s) * vd;
  }
  float xv = x[((size_t)(b * Cc + c)) * Nn + n];
  y1[base + c] = xv + acc;  // residual; y1 kept in (B,N,C)
}

// ---------------------------------------------------------------------------
// Kernel 5/7: BatchNorm batch statistics -> fused scale a[c], shift b[c].
// grid (C), block 256.
// ---------------------------------------------------------------------------
__global__ void bnstats_kernel(const float* __restrict__ y,
                               const float* __restrict__ g,
                               const float* __restrict__ bet,
                               float* __restrict__ aOut,
                               float* __restrict__ bOut) {
  __shared__ float s1[256];
  __shared__ float s2[256];
  const int c = blockIdx.x, tid = threadIdx.x;
  float a = 0.0f, b2 = 0.0f;
  for (int i = tid; i < Bn * Nn; i += 256) {
    float v = y[(size_t)i * Cc + c];
    a += v;
    b2 += v * v;
  }
  s1[tid] = a;
  s2[tid] = b2;
  __syncthreads();
  for (int off = 128; off > 0; off >>= 1) {
    if (tid < off) { s1[tid] += s1[tid + off]; s2[tid] += s2[tid + off]; }
    __syncthreads();
  }
  if (tid == 0) {
    const float inv_n = 1.0f / (float)(Bn * Nn);
    float mean = s1[0] * inv_n;
    float var = s2[0] * inv_n - mean * mean;
    float ia = g[c] * rsqrtf(var + 1e-5f);
    aOut[c] = ia;
    bOut[c] = bet[c] - ia * mean;
  }
}

// ---------------------------------------------------------------------------
// Kernel 6: fused FFN. Per 64-column tile: x1 = bn1(y1) -> LDS (bf16),
// H = LeakyReLU(W1 x1) in LDS (512x64 bf16), y2 = x1 + W2 H. All WMMA.
// grid (N/64, B), block 256 (8 waves). dyn LDS = (64*136 + 64*520)*2 bytes.
// ---------------------------------------------------------------------------
__global__ void ffn_kernel(const float* __restrict__ y1,
                           const float* __restrict__ W1,
                           const float* __restrict__ W2,
                           const float* __restrict__ a1,
                           const float* __restrict__ b1s,
                           float* __restrict__ y2) {
  extern __shared__ char smem[];
  bf16* x1l = (bf16*)smem;                   // [64 n][136] (k=c contiguous)
  bf16* hl = (bf16*)(smem + 64 * 136 * 2);   // [64 n][520] (k=f contiguous)
  const int tid = threadIdx.x;
  const int b = blockIdx.y;
  const int n0 = blockIdx.x * 64;

  for (int e = tid; e < 64 * 128; e += 256) {
    int nl = e >> 7, c = e & 127;
    float v = y1[((size_t)(b * Nn + n0 + nl)) * Cc + c] * a1[c] + b1s[c];
    x1l[nl * 136 + c] = (bf16)v;
  }
  __syncthreads();

  const int wave = tid >> 5, lane = tid & 31;
  const int r = lane & 15, hi = lane >> 4;

  // Matmul 1: H(512x64) = leaky(W1(512x128) @ x1(128x64))
#pragma unroll 1
  for (int mq = 0; mq < 4; ++mq) {
    int mt = wave * 4 + mq;  // f-tile 0..31
    v16bf aw[4];
#pragma unroll
    for (int kt = 0; kt < 4; ++kt)
      aw[kt] = load_a_f32(W1, Cc, mt * 16 + r, hi, kt);
#pragma unroll 1
    for (int nt = 0; nt < 4; ++nt) {
      v8f acc = {0, 0, 0, 0, 0, 0, 0, 0};
#pragma unroll
      for (int kt = 0; kt < 4; ++kt) {
        v16bf bb = load_b_bf16(x1l, 136, nt * 16 + r, hi, kt);
        acc = wmma_bf16(aw[kt], bb, acc);
      }
#pragma unroll
      for (int i = 0; i < 8; ++i) {
        float h = acc[i];
        h = (h >= 0.0f) ? h : 0.2f * h;  // LeakyReLU(0.2)
        hl[(nt * 16 + r) * 520 + mt * 16 + i + 8 * hi] = (bf16)h;
      }
    }
  }
  __syncthreads();

  // Matmul 2: FF(128x64) = W2(128x512) @ H ; y2 = x1 + FF
  {
    int mt = wave;  // co-tile 0..7
    v8f acc4[4];
#pragma unroll
    for (int nt = 0; nt < 4; ++nt) acc4[nt] = (v8f){0, 0, 0, 0, 0, 0, 0, 0};
#pragma unroll 1
    for (int kt = 0; kt < 16; ++kt) {
      v16bf aw = load_a_f32(W2, Ff, mt * 16 + r, hi, kt);
#pragma unroll
      for (int nt = 0; nt < 4; ++nt) {
        v16bf bb = load_b_bf16(hl, 520, nt * 16 + r, hi, kt);
        acc4[nt] = wmma_bf16(aw, bb, acc4[nt]);
      }
    }
#pragma unroll 1
    for (int nt = 0; nt < 4; ++nt) {
#pragma unroll
      for (int i = 0; i < 8; ++i) {
        int co = mt * 16 + i + 8 * hi;
        int n = n0 + nt * 16 + r;
        size_t o = ((size_t)(b * Nn + n)) * Cc + co;
        float x1v = y1[o] * a1[co] + b1s[co];
        y2[o] = x1v + acc4[nt][i];
      }
    }
  }
}

// ---------------------------------------------------------------------------
// Kernel 8: final BN apply + transpose (B,N,C) -> (B,C,N) output layout.
// ---------------------------------------------------------------------------
__global__ void bnapply_kernel(const float* __restrict__ y2,
                               const float* __restrict__ a2,
                               const float* __restrict__ b2s,
                               float* __restrict__ out) {
  int gid = blockIdx.x * 256 + threadIdx.x;  // coalesced over n on write
  int n = gid & (Nn - 1);
  int c = (gid >> 12) & (Cc - 1);
  int b = gid >> 19;
  float v = y2[((size_t)(b * Nn + n)) * Cc + c];
  out[((size_t)(b * Cc + c)) * Nn + n] = a2[c] * v + b2s[c];
}

// ---------------------------------------------------------------------------
extern "C" void kernel_launch(void* const* d_in, const int* in_sizes, int n_in,
                              void* d_out, int out_size, void* d_ws,
                              size_t ws_size, hipStream_t stream) {
  (void)in_sizes; (void)n_in; (void)out_size; (void)ws_size;
  const float* x  = (const float*)d_in[0];
  const float* Wq = (const float*)d_in[1];
  const float* Wk = (const float*)d_in[2];
  const float* Wv = (const float*)d_in[3];
  const float* W1 = (const float*)d_in[4];
  const float* W2 = (const float*)d_in[5];
  const float* g1 = (const float*)d_in[6];
  const float* b1 = (const float*)d_in[7];
  const float* g2 = (const float*)d_in[8];
  const float* b2 = (const float*)d_in[9];

  char* ws = (char*)d_ws;
  size_t off = 0;
  auto alloc = [&](size_t bytes) -> void* {
    off = (off + 255) & ~(size_t)255;
    void* p = ws + off;
    off += bytes;
    return p;
  };
  const size_t fPlane = (size_t)Bn * Nn * Cc * sizeof(float);  // 8 MB
  float* q   = (float*)alloc(fPlane);
  float* kx  = (float*)alloc(fPlane);
  float* vx  = (float*)alloc(fPlane);
  float* y1  = (float*)alloc(fPlane);
  float* y2  = (float*)alloc(fPlane);
  bf16*  xb  = (bf16*)alloc((size_t)Bn * Nn * Cc * sizeof(bf16));
  float* sqn = (float*)alloc((size_t)Bn * Nn * sizeof(float));
  int*   idx = (int*)alloc((size_t)Bn * Nn * Kk * sizeof(int));
  float* a1  = (float*)alloc(Cc * sizeof(float));
  float* b1s = (float*)alloc(Cc * sizeof(float));
  float* a2  = (float*)alloc(Cc * sizeof(float));
  float* b2s = (float*)alloc(Cc * sizeof(float));

  const size_t qkvLds = (size_t)2 * 128 * 132 * sizeof(float);   // 132 KB
  const size_t knnLds = (size_t)16 * Nn * sizeof(float);         // 256 KB
  const size_t ffnLds = (64 * 136 + 64 * 520) * sizeof(bf16);    // 82 KB

  qkv_kernel<<<dim3(Nn / 128, Bn), 256, qkvLds, stream>>>(x, Wq, Wk, Wv, q,
                                                          kx, vx, xb);
  sqnorm_kernel<<<(Bn * Nn) / 256, 256, 0, stream>>>(x, sqn);
  knn_kernel<<<dim3(Nn / 16, Bn), 512, knnLds, stream>>>(xb, sqn, idx);
  attn_kernel<<<dim3(Nn, Bn), 128, 0, stream>>>(x, q, kx, vx, idx, y1);
  bnstats_kernel<<<Cc, 256, 0, stream>>>(y1, g1, b1, a1, b1s);
  ffn_kernel<<<dim3(Nn / 64, Bn), 256, ffnLds, stream>>>(y1, W1, W2, a1, b1s,
                                                         y2);
  bnstats_kernel<<<Cc, 256, 0, stream>>>(y2, g2, b2, a2, b2s);
  bnapply_kernel<<<(size_t)(Bn * Cc * Nn) / 256, 256, 0, stream>>>(y2, a2, b2s,
                                                                   (float*)d_out);
}
